// MultiHeadAttention_54382875902170
// MI455X (gfx1250) — matrix operand order repaired and checked by reading, not compiled, not run on hardware
//
#include <hip/hip_runtime.h>

#define B_  4
#define S_  2048
#define DM  256
#define H_  8
#define HD  32

typedef _Float16 half_t;
typedef __attribute__((ext_vector_type(16))) _Float16 v16h;
typedef __attribute__((ext_vector_type(8)))  float    v8f;

// ds_swizzle_b32, group-of-32 mode: offset = {1'b0, xor[14:10], or[9:5], and[4:0]}
#define DS_SWZ(x, m) __int_as_float(__builtin_amdgcn_ds_swizzle(__float_as_int(x), (((m) << 10) | 0x1f)))

__device__ __forceinline__ float rmax16(float x) {
    x = fmaxf(x, DS_SWZ(x, 1));
    x = fmaxf(x, DS_SWZ(x, 2));
    x = fmaxf(x, DS_SWZ(x, 4));
    x = fmaxf(x, DS_SWZ(x, 8));
    return x;
}
__device__ __forceinline__ float rsum16(float x) {
    x += DS_SWZ(x, 1);
    x += DS_SWZ(x, 2);
    x += DS_SWZ(x, 4);
    x += DS_SWZ(x, 8);
    return x;
}
__device__ __forceinline__ float rsum32(float x) {
    x += DS_SWZ(x, 1);
    x += DS_SWZ(x, 2);
    x += DS_SWZ(x, 4);
    x += DS_SWZ(x, 8);
    x += DS_SWZ(x, 16);
    return x;
}

// ---------------------------------------------------------------------------
// Kernel 0: convert the 4 weight matrices fp32 row-major -> f16 TRANSPOSED
// (wt[n*256 + k] = W[k*256 + n]) so WMMA B-fragments are contiguous 32B loads.
// ---------------------------------------------------------------------------
__global__ void wconv_kernel(const float* __restrict__ Wq, const float* __restrict__ Wk,
                             const float* __restrict__ Wv, const float* __restrict__ Wo,
                             half_t* __restrict__ out) {
    int id  = blockIdx.x * blockDim.x + threadIdx.x;   // 4*65536 total, exact grid
    int mat = id >> 16;
    int rem = id & 0xFFFF;
    int k   = rem >> 8;
    int n   = rem & 255;
    const float* W = (mat == 0) ? Wq : (mat == 1) ? Wk : (mat == 2) ? Wv : Wo;
    out[(size_t)mat * 65536 + n * 256 + k] = (half_t)W[k * 256 + n];
}

// ---------------------------------------------------------------------------
// Kernel 1: fused Q/K/V projection GEMMs (M=8192, N=256, K=256), f16 output.
// One 16x16 output tile per wave, K-loop of 8 x v_wmma_f32_16x16x32_f16.
// blockIdx.z selects which projection (0=q,1=k,2=v).
// ---------------------------------------------------------------------------
__global__ __launch_bounds__(256) void proj_kernel(
        const float* __restrict__ Qin, const float* __restrict__ Kin, const float* __restrict__ Vin,
        const half_t* __restrict__ wt_all,
        const float* __restrict__ bqp, const float* __restrict__ bkp, const float* __restrict__ bvp,
        half_t* __restrict__ qo, half_t* __restrict__ ko, half_t* __restrict__ vo) {
    const int z = blockIdx.z;
    const float*  A    = (z == 0) ? Qin : (z == 1) ? Kin : Vin;
    const half_t* Wt   = wt_all + (size_t)z * 65536;
    const float*  bias = (z == 0) ? bqp : (z == 1) ? bkp : bvp;
    half_t*       Out  = (z == 0) ? qo : (z == 1) ? ko : vo;

    const int lane = threadIdx.x & 31;
    const int wave = threadIdx.x >> 5;
    const int rowbase = blockIdx.x * 16;
    const int nbase   = blockIdx.y * 128 + wave * 16;
    const int m  = lane & 15;      // A row within tile / B column within tile
    const int hs = lane >> 4;      // half-wave select

    v8f c = {0.f, 0.f, 0.f, 0.f, 0.f, 0.f, 0.f, 0.f};
    const float*  arow = A  + (size_t)(rowbase + m) * DM;
    const half_t* wcol = Wt + (size_t)(nbase + m) * DM + hs * 16;

    for (int kk = 0; kk < DM; kk += 32) {
        // A fragment (16x32 f16): contiguous k = 16*hs + s (matches B)
        v16h a;
        const float* p0 = arow + kk + hs * 16;
        const float* p1 = arow + kk + hs * 16 + 8;
#pragma unroll
        for (int t = 0; t < 8; ++t) {
            a[t]     = (half_t)p0[t];
            a[8 + t] = (half_t)p1[t];
        }
        // B fragment (32x16 f16): contiguous 16 halves from transposed weights
        v16h bfrag = *(const v16h*)(wcol + kk);
        c = __builtin_amdgcn_wmma_f32_16x16x32_f16(false, a, false, bfrag, (short)0, c, false, false);
    }

    const float bvv = bias[nbase + m];
#pragma unroll
    for (int r = 0; r < 8; ++r) {
        int row = rowbase + hs * 8 + r;
        Out[(size_t)row * DM + nbase + m] = (half_t)(c[r] + bvv);
    }
}

// ---------------------------------------------------------------------------
// Kernel 2: flash attention with per-head exponential decay + causal mask.
// Block = 8 waves = 128 query rows of one (batch, head). Per 32-key block:
// all 256 threads cooperatively stage K (2KB) + V (2KB) into LDS with ONE
// global_load_async_to_lds_b128 per wave, synchronized with
// s_wait_asynccnt + workgroup barrier. Each wave then runs 2 score WMMAs +
// online softmax (ds_swizzle row reductions) + 2 P@V WMMAs, with P
// re-laid-out C->A through a per-wave LDS staging buffer.
// ---------------------------------------------------------------------------
__global__ __launch_bounds__(256) void attn_kernel(
        const half_t* __restrict__ q16, const half_t* __restrict__ k16,
        const half_t* __restrict__ v16, const float* __restrict__ gammas,
        float* __restrict__ O) {
    __shared__ half_t Kbuf[32 * 32];    // [key][hd], 2KB
    __shared__ half_t Vbuf[32 * 32];    // [key][hd], 2KB
    __shared__ half_t Pl[8][16 * 32];   // per-wave P staging, 8KB

    const int lane = threadIdx.x & 31;
    const int wave = threadIdx.x >> 5;
    const int b = blockIdx.z, h = blockIdx.y;
    const int qblock = blockIdx.x * 128;
    const int qbase  = qblock + wave * 16;
    const int m  = lane & 15;
    const int hs = lane >> 4;
    const float gamma = gammas[h];
    const float scale = 0.17677669529663687f;   // 1/sqrt(32)

    // --- cooperative staging map: 256 threads x 16B = 4KB (K block + V block)
    // waves 0..3 fill Kbuf, waves 4..7 fill Vbuf (wave-uniform source pointer)
    const int  obyte = threadIdx.x * 16;              // 0..4080
    const bool isK   = obyte < 2048;
    const int  oo    = isK ? obyte : (obyte - 2048);  // offset inside K or V block
    const int  krow  = oo >> 6;                       // 64 bytes per key row
    const int  kcol  = oo & 63;
    const half_t* gsrc = isK ? k16 : v16;
    const unsigned goff0 =
        (unsigned)(((b * S_ + krow) * DM + h * HD) * 2 + kcol);   // += kv*512 per block
    const unsigned ldsdst =
        (unsigned)(unsigned long long)(isK ? &Kbuf[0] : &Vbuf[0]) + (unsigned)oo;

    // Q fragment (16 rows x 32 head-dim), loaded once (K = head_dim = 32)
    const half_t* qrow = q16 + ((size_t)(b * S_) + qbase + m) * DM + h * HD;
    v16h qa;
#pragma unroll
    for (int t = 0; t < 8; ++t) {
        qa[t]     = qrow[hs * 16 + t];
        qa[8 + t] = qrow[hs * 16 + 8 + t];
    }

    v8f acc0 = {0.f, 0.f, 0.f, 0.f, 0.f, 0.f, 0.f, 0.f};  // O cols 0..15
    v8f acc1 = {0.f, 0.f, 0.f, 0.f, 0.f, 0.f, 0.f, 0.f};  // O cols 16..31
    float mrow[8], lrow[8];
#pragma unroll
    for (int r = 0; r < 8; ++r) { mrow[r] = -3.0e38f; lrow[r] = 0.f; }

    const int myEnd    = qbase + 15;     // this wave's causal limit
    const int blockEnd = qblock + 127;   // uniform trip count for the block

    for (int kv = 0; kv <= blockEnd; kv += 32) {
        // ---- async stage K/V block into LDS (one async b128 per wave)
        unsigned goff = goff0 + (unsigned)kv * (DM * 2);
        asm volatile("global_load_async_to_lds_b128 %0, %1, %2"
                     :: "v"(ldsdst), "v"(goff), "s"(gsrc) : "memory");
        asm volatile("s_wait_asynccnt 0" ::: "memory");
        __syncthreads();

        if (kv <= myEnd) {   // wave-uniform branch: EXEC stays all-1s for WMMA
            // K^T B-fragments straight from LDS (2 x ds_load_b128 each)
            v16h kb0 = *(const v16h*)&Kbuf[m * 32 + hs * 16];
            v16h kb1 = *(const v16h*)&Kbuf[(16 + m) * 32 + hs * 16];

            v8f zero = {0.f, 0.f, 0.f, 0.f, 0.f, 0.f, 0.f, 0.f};
            v8f s0 = __builtin_amdgcn_wmma_f32_16x16x32_f16(false, qa, false, kb0, (short)0, zero, false, false);
            v8f s1 = __builtin_amdgcn_wmma_f32_16x16x32_f16(false, qa, false, kb1, (short)0, zero, false, false);

            // online softmax with decay: logit = (j<=i) ? s*scale*exp(-g*(i-j)) : -1e4
#pragma unroll
            for (int r = 0; r < 8; ++r) {
                const int i  = qbase + hs * 8 + r;
                const int d0 = i - (kv + m);
                const int d1 = i - (kv + 16 + m);
                float l0 = (d0 >= 0) ? s0[r] * scale * __expf(-gamma * (float)d0) : -10000.f;
                float l1 = (d1 >= 0) ? s1[r] * scale * __expf(-gamma * (float)d1) : -10000.f;

                float bm = rmax16(fmaxf(l0, l1));     // row max over this 32-key block
                float mn = fmaxf(mrow[r], bm);
                float alpha = __expf(mrow[r] - mn);
                float p0 = __expf(l0 - mn);
                float p1 = __expf(l1 - mn);
                float rs = rsum16(p0 + p1);           // row sum over this block
                lrow[r] = lrow[r] * alpha + rs;
                mrow[r] = mn;
                acc0[r] *= alpha;
                acc1[r] *= alpha;

                const int mm = hs * 8 + r;            // C-layout -> LDS (m, j)
                Pl[wave][mm * 32 + m]      = (half_t)p0;
                Pl[wave][mm * 32 + 16 + m] = (half_t)p1;
            }
            // cross-lane LDS RAW within the wave: explicit DS-counter wait
            asm volatile("s_wait_dscnt 0" ::: "memory");

            // P as A-fragment (16 rows x 32 keys)
            v16h pa;
#pragma unroll
            for (int t = 0; t < 8; ++t) {
                pa[t]     = Pl[wave][m * 32 + hs * 16 + t];
                pa[8 + t] = Pl[wave][m * 32 + hs * 16 + 8 + t];
            }
            // V B-fragments gathered from LDS (32 key-positions x 16 hd cols)
            v16h vb0, vb1;
#pragma unroll
            for (int t = 0; t < 16; ++t) {
                vb0[t] = Vbuf[(hs * 16 + t) * 32 + m];
                vb1[t] = Vbuf[(hs * 16 + t) * 32 + 16 + m];
            }
            acc0 = __builtin_amdgcn_wmma_f32_16x16x32_f16(false, pa, false, vb0, (short)0, acc0, false, false);
            acc1 = __builtin_amdgcn_wmma_f32_16x16x32_f16(false, pa, false, vb1, (short)0, acc1, false, false);
        }
        __syncthreads();   // protect Kbuf/Vbuf before next block's async writes
    }

#pragma unroll
    for (int r = 0; r < 8; ++r) {
        float inv = 1.0f / lrow[r];
        int row = qbase + hs * 8 + r;
        float* orow = O + ((size_t)(b * S_) + row) * DM + h * HD;
        orow[m]      = acc0[r] * inv;
        orow[16 + m] = acc1[r] * inv;
    }
}

// ---------------------------------------------------------------------------
// Kernel 3: output projection + residual:  X = Qin + O @ Wo + bo  (fp32 out)
// ---------------------------------------------------------------------------
__global__ __launch_bounds__(256) void oproj_kernel(
        const float* __restrict__ Oin, const half_t* __restrict__ Wt,
        const float* __restrict__ bo, const float* __restrict__ Qres,
        float* __restrict__ X) {
    const int lane = threadIdx.x & 31;
    const int wave = threadIdx.x >> 5;
    const int rowbase = blockIdx.x * 16;
    const int nbase   = blockIdx.y * 128 + wave * 16;
    const int m  = lane & 15;
    const int hs = lane >> 4;

    v8f c = {0.f, 0.f, 0.f, 0.f, 0.f, 0.f, 0.f, 0.f};
    const float*  arow = Oin + (size_t)(rowbase + m) * DM;
    const half_t* wcol = Wt  + (size_t)(nbase + m) * DM + hs * 16;

    for (int kk = 0; kk < DM; kk += 32) {
        v16h a;
        const float* p0 = arow + kk + hs * 16;
        const float* p1 = arow + kk + hs * 16 + 8;
#pragma unroll
        for (int t = 0; t < 8; ++t) {
            a[t]     = (half_t)p0[t];
            a[8 + t] = (half_t)p1[t];
        }
        v16h bfrag = *(const v16h*)(wcol + kk);
        c = __builtin_amdgcn_wmma_f32_16x16x32_f16(false, a, false, bfrag, (short)0, c, false, false);
    }

    const float bvv = bo[nbase + m];
#pragma unroll
    for (int r = 0; r < 8; ++r) {
        size_t idx = (size_t)(rowbase + hs * 8 + r) * DM + nbase + m;
        X[idx] = Qres[idx] + c[r] + bvv;
    }
}

// ---------------------------------------------------------------------------
// Kernel 4: row LayerNorm over 256 columns (one block per row, 8 waves)
// ---------------------------------------------------------------------------
__global__ __launch_bounds__(256) void ln_kernel(
        const float* __restrict__ X, const float* __restrict__ g,
        const float* __restrict__ be, float* __restrict__ out) {
    __shared__ float red[8];
    const int row = blockIdx.x;
    const int t   = threadIdx.x;
    float x = X[(size_t)row * DM + t];

    float s = rsum32(x);
    if ((t & 31) == 0) red[t >> 5] = s;
    __syncthreads();
    float tot = 0.f;
#pragma unroll
    for (int i = 0; i < 8; ++i) tot += red[i];
    const float mu = tot * (1.0f / 256.0f);
    const float d  = x - mu;
    __syncthreads();

    float s2 = rsum32(d * d);
    if ((t & 31) == 0) red[t >> 5] = s2;
    __syncthreads();
    float tv = 0.f;
#pragma unroll
    for (int i = 0; i < 8; ++i) tv += red[i];
    const float var = tv * (1.0f / 256.0f);

    out[(size_t)row * DM + t] = d * rsqrtf(var + 1e-5f) * g[t] + be[t];
}

// ---------------------------------------------------------------------------
extern "C" void kernel_launch(void* const* d_in, const int* in_sizes, int n_in,
                              void* d_out, int out_size, void* d_ws, size_t ws_size,
                              hipStream_t stream) {
    const float* Q   = (const float*)d_in[0];
    const float* K   = (const float*)d_in[1];
    const float* V   = (const float*)d_in[2];
    /* d_in[3] = causal mask (int), implemented analytically */
    const float* gam = (const float*)d_in[4];
    const float* Wq  = (const float*)d_in[5];
    const float* bq  = (const float*)d_in[6];
    const float* Wk  = (const float*)d_in[7];
    const float* bk  = (const float*)d_in[8];
    const float* Wv  = (const float*)d_in[9];
    const float* bv  = (const float*)d_in[10];
    const float* Wo  = (const float*)d_in[11];
    const float* bo  = (const float*)d_in[12];
    const float* lng = (const float*)d_in[13];
    const float* lnb = (const float*)d_in[14];

    // workspace carve-up (all offsets 32B-aligned)
    char* ws = (char*)d_ws;
    half_t* wt  = (half_t*)ws;                               // 4 * 65536 f16  = 512 KB
    half_t* q16 = (half_t*)(ws + 524288);                    // 8192*256 f16   = 4 MB
    half_t* k16 = q16 + (size_t)8192 * 256;
    half_t* v16 = k16 + (size_t)8192 * 256;
    float*  O32 = (float*)(ws + 524288 + 3u * 4194304u);     // 8192*256 f32   = 8 MB
    float*  X32 = O32 + (size_t)8192 * 256;                  // 8192*256 f32   = 8 MB

    wconv_kernel<<<1024, 256, 0, stream>>>(Wq, Wk, Wv, Wo, wt);
    proj_kernel<<<dim3(512, 2, 3), 256, 0, stream>>>(Q, K, V, wt, bq, bk, bv, q16, k16, v16);
    attn_kernel<<<dim3(S_ / 128, H_, B_), 256, 0, stream>>>(q16, k16, v16, gam, O32);
    oproj_kernel<<<dim3(512, 2, 1), 256, 0, stream>>>(O32, wt + (size_t)3 * 65536, bo, Q, X32);
    ln_kernel<<<8192, 256, 0, stream>>>(X32, lng, lnb, (float*)d_out);
}
